// Encoder_Recovery_43422119362859
// MI455X (gfx1250) — compile-verified, matrix operand order checked by reading, not run on hardware
//
#include <hip/hip_runtime.h>

// ---------------------------------------------------------------------------
// MI455X (gfx1250) implementation: all 3x3 convs as implicit GEMM on
// v_wmma_f32_16x16x32_f16 (wave32). f16 activations/weights, fp32 accum.
// Block tile 64(cout) x 256(pixels), 8 wave32 (2x4), 2x4 16x16 frags per wave,
// double-buffered LDS (one barrier per K step), bank-conflict-free 40-f16 row
// stride, packed b128 epilogue stores.
// Workspace layout (assumes ws_size >= ~285 MB):
//   [0, 128MiB)        actA  (f16 ping)
//   [128MiB, 256MiB)   actB  (f16 pong)
//   [256MiB, +10MiB)   pre-transformed f16 weights, all 8 layers
//   [+12MiB, +8KiB)    BN stats: sum[512], sumsq[512], scale/shift float2[512]
// ---------------------------------------------------------------------------

typedef _Float16 h8  __attribute__((ext_vector_type(8)));
typedef _Float16 h16 __attribute__((ext_vector_type(16)));
typedef float    f8  __attribute__((ext_vector_type(8)));

#define DEVINL __device__ __forceinline__

// LDS row stride (f16) for A/B tiles: 32 payload + 8 pad -> 80B row pitch,
// start-bank step 20 (gcd(20,64)=4, cycle 16) => fragment ds_load_b128s from
// 16 lanes hit 16 distinct 4-bank groups (conflict-free).
#define LPITCH 40

// Load a 16x32 f16 A/B fragment slice from LDS, matching the CDNA5 16-bit
// operand lane layout: lanes 0-15 hold K=0..7 & 16..23 of row (lane),
// lanes 16-31 hold K=8..15 & 24..31 of row (lane-16). rowbase = &tile[row*LPITCH].
DEVINL h16 frag_ld(const _Float16* rowbase, int lane) {
  const _Float16* p = rowbase + ((lane & 16) ? 8 : 0);
  h8 lo = *(const h8*)(p);
  h8 hi = *(const h8*)(p + 16);
  return __builtin_shufflevector(lo, hi, 0,1,2,3,4,5,6,7,8,9,10,11,12,13,14,15);
}

// ---------------------------------------------------------------------------
// Implicit-GEMM 3x3 conv (pad=1) over NHWC f16, N=256 images of HWxHW.
// K loop steps 32 over (kh,kw,channel), channel-innermost (CIN % 32 == 0) so
// every chunk is 32 contiguous channels of one shifted pixel.
// Epilogue: packed h8 pre-BN stores + per-channel sum/sumsq for training BN.
// ---------------------------------------------------------------------------
template<int CIN, int COUT, int HW>
__global__ __launch_bounds__(256)
void conv3x3_wmma(const _Float16* __restrict__ X,   // [256*HW*HW][CIN]
                  const _Float16* __restrict__ Wk,  // [COUT][9*CIN]  k=(kh*3+kw)*CIN+c
                  _Float16* __restrict__ Y,         // [256*HW*HW][COUT] pre-BN
                  float* __restrict__ gSum, float* __restrict__ gSq)
{
  constexpr int PIX  = HW * HW;
  constexpr int KTOT = 9 * CIN;
  constexpr int TN = 256, TM = 64, BK = 32;

  __shared__ _Float16 lA[2][TM * LPITCH];   // 2 x 5 KB
  __shared__ _Float16 lB[2][TN * LPITCH];   // 2 x 20 KB
  __shared__ float redS[TM];
  __shared__ float redQ[TM];

  const int tid  = threadIdx.x;
  const int lane = tid & 31;
  const int wave = tid >> 5;
  const int wm   = wave >> 2;       // 0..1  (M slab of 32)
  const int wn   = wave & 3;        // 0..3  (N slab of 64)

  // cooperative B load: thread t fetches all 32 channels of pixel t (64B)
  const int gp    = blockIdx.x * TN + tid;
  const int n_img = gp / PIX;
  const int rem   = gp % PIX;
  const int oh    = rem / HW;
  const int ow    = rem % HW;
  const _Float16* xPix = X + (size_t)n_img * PIX * CIN;

  // cooperative A load: thread -> (row, 8 contiguous k)
  const int aRow = tid >> 2;              // 0..63
  const int aKo  = (tid & 3) * 8;
  const _Float16* wRow = Wk + (size_t)(blockIdx.y * TM + aRow) * KTOT;

  const f8 zero8 = {0.f,0.f,0.f,0.f,0.f,0.f,0.f,0.f};
  f8 acc[2][4];
  #pragma unroll
  for (int fm = 0; fm < 2; ++fm)
    #pragma unroll
    for (int fn = 0; fn < 4; ++fn) acc[fm][fn] = zero8;
  const h8 hzero = {0,0,0,0,0,0,0,0};

  h8 nb[4], na;

  // ---- fetch K-chunk `ks` into registers (B: 32ch of shifted pixel, A: 8 wts)
  auto fetch = [&](int ks, h8 nbv[4], h8& nav) {
    const int t  = ks / CIN;
    const int c0 = ks % CIN;
    const int kh = t / 3, kw = t % 3;
    const int ih = oh + kh - 1;
    const int iw = ow + kw - 1;
    nbv[0] = hzero; nbv[1] = hzero; nbv[2] = hzero; nbv[3] = hzero;
    if ((ih >= 0) & (ih < HW) & (iw >= 0) & (iw < HW)) {
      const _Float16* src = xPix + (size_t)(ih * HW + iw) * CIN + c0;
      nbv[0] = *(const h8*)(src);
      nbv[1] = *(const h8*)(src + 8);
      nbv[2] = *(const h8*)(src + 16);
      nbv[3] = *(const h8*)(src + 24);
    }
    nav = *(const h8*)(wRow + ks + aKo);
  };
  auto stage = [&](int buf, const h8 nbv[4], const h8 nav) {
    _Float16* bd = &lB[buf][tid * LPITCH];
    *(h8*)(bd)      = nbv[0];
    *(h8*)(bd + 8)  = nbv[1];
    *(h8*)(bd + 16) = nbv[2];
    *(h8*)(bd + 24) = nbv[3];
    *(h8*)(&lA[buf][aRow * LPITCH + aKo]) = nav;
  };

  // prologue: stage K-chunk 0 into buffer 0
  fetch(0, nb, na);
  stage(0, nb, na);
  __syncthreads();

  int buf = 0;
  for (int ks = 0; ks < KTOT; ks += BK) {
    const bool have_next = (ks + BK) < KTOT;
    if (have_next) {
      fetch(ks + BK, nb, na);                     // overlap with compute below
      __builtin_prefetch(wRow + ks + 2 * BK + aKo, 0, 0);
    }

    // compute from LDS[buf]
    h16 afr[2], bfr[4];
    #pragma unroll
    for (int fm = 0; fm < 2; ++fm) {
      const int row = wm * 32 + fm * 16 + (lane & 15);
      afr[fm] = frag_ld(&lA[buf][row * LPITCH], lane);
    }
    #pragma unroll
    for (int fn = 0; fn < 4; ++fn) {
      const int col = wn * 64 + fn * 16 + (lane & 15);
      bfr[fn] = frag_ld(&lB[buf][col * LPITCH], lane);
    }
    #pragma unroll
    for (int fm = 0; fm < 2; ++fm)
      #pragma unroll
      for (int fn = 0; fn < 4; ++fn)
        acc[fm][fn] = __builtin_amdgcn_wmma_f32_16x16x32_f16(
            false, afr[fm], false, bfr[fn], (short)0, acc[fm][fn], false, false);

    if (have_next) stage(buf ^ 1, nb, na);        // other buffer: no race
    __syncthreads();                              // reads(buf) + writes(buf^1) done
    buf ^= 1;
  }

  // ---- epilogue: packed pre-BN f16 stores + per-channel batch statistics ----
  if (tid < TM) { redS[tid] = 0.f; redQ[tid] = 0.f; }
  __syncthreads();

  #pragma unroll
  for (int fm = 0; fm < 2; ++fm) {
    // C/D layout: VGPR r holds M=r (lanes 0-15) and M=r+8 (lanes 16-31),
    // so each lane's 8 accum values are 8 *consecutive* couts -> one b128 store.
    const int mbase = wm * 32 + fm * 16 + ((lane & 16) ? 8 : 0);
    float s[8], q[8];
    #pragma unroll
    for (int r = 0; r < 8; ++r) { s[r] = 0.f; q[r] = 0.f; }
    #pragma unroll
    for (int fn = 0; fn < 4; ++fn) {
      const int col  = wn * 64 + fn * 16 + (lane & 15);
      const int gpix = blockIdx.x * TN + col;
      h8 o;
      #pragma unroll
      for (int r = 0; r < 8; ++r) {
        const float v = acc[fm][fn][r];
        o[r] = (_Float16)v;
        s[r] += v;
        q[r] += v * v;
      }
      *(h8*)(&Y[(size_t)gpix * COUT + blockIdx.y * TM + mbase]) = o;
    }
    #pragma unroll
    for (int r = 0; r < 8; ++r) {
      atomicAdd(&redS[mbase + r], s[r]);
      atomicAdd(&redQ[mbase + r], q[r]);
    }
  }
  __syncthreads();
  if (tid < TM) {
    atomicAdd(&gSum[blockIdx.y * TM + tid], redS[tid]);
    atomicAdd(&gSq [blockIdx.y * TM + tid], redQ[tid]);
  }
}

// ---------------------------------------------------------------------------
// Gather K=32 blocks of 64x64 per image out of NCHW fp32 input; emit NHWC f16
// with channels zero-padded 3 -> 32 so layer 1 also runs on the WMMA path.
// ---------------------------------------------------------------------------
__global__ void gather_blocks(const float* __restrict__ x, const int* __restrict__ bids,
                              _Float16* __restrict__ r0)
{
  const int idx = blockIdx.x * blockDim.x + threadIdx.x;  // (b*K+ki, oh, ow)
  if (idx >= 256 * 64 * 64) return;
  const int ow = idx & 63;
  const int oh = (idx >> 6) & 63;
  const int nb = idx >> 12;
  const int b  = nb >> 5;
  const int ki = nb & 31;
  const int g  = bids[ki];
  const int h  = (g >> 3) * 64 + oh;   // NB blocks laid out as (H/64, W/64)
  const int w  = (g & 7) * 64 + ow;
  const size_t base = (size_t)b * 3 * 512 * 512 + (size_t)h * 512 + w;
  h8 o0 = {0,0,0,0,0,0,0,0};
  o0[0] = (_Float16)x[base];
  o0[1] = (_Float16)x[base + 512 * 512];
  o0[2] = (_Float16)x[base + 2 * 512 * 512];
  const h8 z = {0,0,0,0,0,0,0,0};
  _Float16* dst = r0 + (size_t)idx * 32;
  *(h8*)(dst)      = o0;
  *(h8*)(dst + 8)  = z;
  *(h8*)(dst + 16) = z;
  *(h8*)(dst + 24) = z;
}

// OIHW fp32 -> [Cout][9*CinP] f16 with k=(kh*3+kw)*CinP + c (zero-pad c>=Cin)
__global__ void prep_weights(const float* __restrict__ w, _Float16* __restrict__ o,
                             int Cout, int Cin, int CinP)
{
  const int i  = blockIdx.x * blockDim.x + threadIdx.x;
  const int KT = 9 * CinP;
  if (i >= Cout * KT) return;
  const int co = i / KT;
  const int k  = i % KT;
  const int t  = k / CinP;
  const int c  = k % CinP;
  const int kh = t / 3, kw = t % 3;
  const float v = (c < Cin) ? w[((size_t)(co * Cin + c) * 3 + kh) * 3 + kw] : 0.f;
  o[i] = (_Float16)v;
}

// Batch stats -> per-channel scale/shift (training-mode BN, biased var, eps=1e-5)
__global__ void bn_finalize(const float* __restrict__ sum, const float* __restrict__ sq,
                            const float* __restrict__ g, const float* __restrict__ b,
                            float2* __restrict__ ss, int C, float invCnt)
{
  const int c = blockIdx.x * blockDim.x + threadIdx.x;
  if (c >= C) return;
  const float m  = sum[c] * invCnt;
  const float v  = sq[c] * invCnt - m * m;
  const float sc = g[c] * rsqrtf(v + 1e-5f);
  ss[c] = make_float2(sc, b[c] - m * sc);
}

// x = relu(y*scale + shift), elementwise over [pix][C] f16, 8 channels/thread
__global__ void bn_relu_apply(const _Float16* __restrict__ y, const float2* __restrict__ ss,
                              _Float16* __restrict__ xo, int C, size_t total)
{
  const size_t i = ((size_t)blockIdx.x * blockDim.x + threadIdx.x) * 8;
  if (i >= total) return;
  const int c = (int)(i % C);
  h8 v = *(const h8*)(y + i);
  h8 o;
  #pragma unroll
  for (int j = 0; j < 8; ++j) {
    const float2 s = ss[c + j];
    o[j] = (_Float16)fmaxf((float)v[j] * s.x + s.y, 0.f);
  }
  *(h8*)(xo + i) = o;
}

// x = maxpool2x2(relu(y*scale + shift)); y is (N, 2HWo, 2HWo, C), out (N,HWo,HWo,C)
__global__ void bn_relu_pool_k(const _Float16* __restrict__ y, const float2* __restrict__ ss,
                               _Float16* __restrict__ xo, int C, int HWo, int total)
{
  const int i = blockIdx.x * blockDim.x + threadIdx.x;
  if (i >= total) return;
  const int c8 = C >> 3;
  const int cg = i % c8;
  const int p  = i / c8;
  const int c  = cg << 3;
  const int HWi = HWo * 2;
  const int ow = p % HWo;
  int t = p / HWo;
  const int oh = t % HWo;
  const int n  = t / HWo;
  float2 s[8];
  #pragma unroll
  for (int j = 0; j < 8; ++j) s[j] = ss[c + j];
  float m[8];
  #pragma unroll
  for (int j = 0; j < 8; ++j) m[j] = 0.f;   // relu >= 0, so 0 is a valid floor
  #pragma unroll
  for (int dy = 0; dy < 2; ++dy)
    #pragma unroll
    for (int dx = 0; dx < 2; ++dx) {
      const size_t off = ((size_t)((n * HWi + (oh * 2 + dy)) * HWi + (ow * 2 + dx))) * C + c;
      h8 v = *(const h8*)(y + off);
      #pragma unroll
      for (int j = 0; j < 8; ++j)
        m[j] = fmaxf(m[j], fmaxf((float)v[j] * s[j].x + s[j].y, 0.f));
    }
  h8 o;
  #pragma unroll
  for (int j = 0; j < 8; ++j) o[j] = (_Float16)m[j];
  *(h8*)(xo + (size_t)p * C + c) = o;
}

// 1x1 conv 512->2 + bias + sigmoid; out in NCHW fp32 (256,2,4,4)
__global__ void final_head(const _Float16* __restrict__ p4, const float* __restrict__ lw,
                           const float* __restrict__ lb, float* __restrict__ out)
{
  const int i = blockIdx.x * blockDim.x + threadIdx.x;   // 8192
  if (i >= 256 * 2 * 16) return;
  const int pix = i & 15;
  const int o   = (i >> 4) & 1;
  const int n   = i >> 5;
  const _Float16* src = p4 + ((size_t)n * 16 + pix) * 512;
  const float* w = lw + o * 512;
  float s = 0.f;
  for (int c = 0; c < 512; ++c) s += (float)src[c] * w[c];
  s += lb[o];
  out[i] = 1.f / (1.f + __expf(-s));
}

// ---------------------------------------------------------------------------
extern "C" void kernel_launch(void* const* d_in, const int* in_sizes, int n_in,
                              void* d_out, int out_size, void* d_ws, size_t ws_size,
                              hipStream_t stream)
{
  (void)in_sizes; (void)n_in; (void)out_size; (void)ws_size;
  const float* x    = (const float*)d_in[0];
  const int*   bids = (const int*)  d_in[1];
  const float* Wf[8]; const float* Gf[8]; const float* Bf[8];
  for (int l = 0; l < 8; ++l) {
    Wf[l] = (const float*)d_in[2 + 3 * l];
    Gf[l] = (const float*)d_in[3 + 3 * l];
    Bf[l] = (const float*)d_in[4 + 3 * l];
  }
  const float* lw = (const float*)d_in[26];
  const float* lb = (const float*)d_in[27];
  float* out = (float*)d_out;

  char* ws = (char*)d_ws;
  const size_t ACT = 134217728ull;                 // 128 MiB per activation buffer
  _Float16* actA = (_Float16*)ws;
  _Float16* actB = (_Float16*)(ws + ACT);
  _Float16* wts  = (_Float16*)(ws + 2 * ACT);
  float*  sumB = (float*)(ws + 2 * ACT + 12ull * 1024 * 1024);
  float*  sqB  = sumB + 512;
  float2* ssB  = (float2*)(sumB + 1024);

  static const int CINa [8] = {  3, 64, 64,128,128,256,256,512};
  static const int CINP [8] = { 32, 64, 64,128,128,256,256,512};
  static const int COUTa[8] = { 64, 64,128,128,256,256,512,512};
  size_t woff[8]; size_t acc = 0;
  for (int l = 0; l < 8; ++l) { woff[l] = acc; acc += (size_t)COUTa[l] * 9 * CINP[l]; }

  // 1) weight pre-transform (fp32 OIHW -> f16 [Cout][9*CinP])
  for (int l = 0; l < 8; ++l) {
    const int tot = COUTa[l] * 9 * CINP[l];
    prep_weights<<<(tot + 255) / 256, 256, 0, stream>>>(Wf[l], wts + woff[l],
                                                        COUTa[l], CINa[l], CINP[l]);
  }
  // 2) block gather -> actA (NHWC f16, C padded to 32)
  gather_blocks<<<(256 * 64 * 64 + 255) / 256, 256, 0, stream>>>(x, bids, actA);

#define RUN_CONV(CINv, COUTv, HWv, Lidx)                                           \
  do {                                                                             \
    hipMemsetAsync(sumB, 0, (COUTv) * sizeof(float), stream);                      \
    hipMemsetAsync(sqB,  0, (COUTv) * sizeof(float), stream);                      \
    dim3 g((256 * (HWv) * (HWv)) / 256, (COUTv) / 64);                             \
    conv3x3_wmma<CINv, COUTv, HWv><<<g, 256, 0, stream>>>(                         \
        actA, wts + woff[Lidx], actB, sumB, sqB);                                  \
    bn_finalize<<<((COUTv) + 63) / 64, 64, 0, stream>>>(                           \
        sumB, sqB, Gf[Lidx], Bf[Lidx], ssB, (COUTv),                               \
        1.0f / (float)(256 * (HWv) * (HWv)));                                      \
  } while (0)

#define RUN_RELU(COUTv, HWv)                                                       \
  do {                                                                             \
    const size_t tot = (size_t)256 * (HWv) * (HWv) * (COUTv);                      \
    const size_t thr = tot / 8;                                                    \
    bn_relu_apply<<<(unsigned)((thr + 255) / 256), 256, 0, stream>>>(              \
        actB, ssB, actA, (COUTv), tot);                                            \
  } while (0)

#define RUN_POOL(COUTv, HWo)                                                       \
  do {                                                                             \
    const int tot = 256 * (HWo) * (HWo) * ((COUTv) / 8);                           \
    bn_relu_pool_k<<<(tot + 255) / 256, 256, 0, stream>>>(                         \
        actB, ssB, actA, (COUTv), (HWo), tot);                                     \
  } while (0)

  // stage 1: 64x64, 32(pad)->64, 64->64, pool -> 32x32
  RUN_CONV( 32,  64, 64, 0); RUN_RELU( 64, 64);
  RUN_CONV( 64,  64, 64, 1); RUN_POOL( 64, 32);
  // stage 2: 32x32, 64->128, 128->128, pool -> 16x16
  RUN_CONV( 64, 128, 32, 2); RUN_RELU(128, 32);
  RUN_CONV(128, 128, 32, 3); RUN_POOL(128, 16);
  // stage 3: 16x16, 128->256, 256->256, pool -> 8x8
  RUN_CONV(128, 256, 16, 4); RUN_RELU(256, 16);
  RUN_CONV(256, 256, 16, 5); RUN_POOL(256,  8);
  // stage 4: 8x8, 256->512, 512->512, pool -> 4x4
  RUN_CONV(256, 512,  8, 6); RUN_RELU(512,  8);
  RUN_CONV(512, 512,  8, 7); RUN_POOL(512,  4);

  // head: 1x1 conv 512->2 + sigmoid, NCHW fp32 out
  final_head<<<(256 * 2 * 16) / 256, 256, 0, stream>>>(actA, lw, lb, out);

#undef RUN_CONV
#undef RUN_RELU
#undef RUN_POOL
}
